// BlockALiBi_49194555408427
// MI455X (gfx1250) — matrix operand order repaired and checked by reading, not compiled, not run on hardware
//
#include <hip/hip_runtime.h>
#include <hip/hip_bf16.h>
#include <math.h>

// ---------------------------------------------------------------------------
// Transformer block (pre-LN, ALiBi attention) for MI455X / gfx1250, wave32.
// B=4, T=2048, D=512, H=8, HS=64.
// All GEMMs: f16 inputs, f32 accumulate via v_wmma_f32_16x16x32_f16.
// Attention is fused (scores -> softmax -> P@V) with scores held in LDS
// (16 rows x 2048 cols x f32 = 128KB; CDNA5 has 320KB LDS per WGP).
// Attention probabilities (537MB) are written once with non-temporal stores.
// ---------------------------------------------------------------------------

typedef __attribute__((ext_vector_type(16))) _Float16 v16h;
typedef __attribute__((ext_vector_type(8)))  float    v8f;

struct FragH {
    union {
        v16h h;
        uint4 q[2];
        _Float16 e[16];
    };
};

#define CB 4
#define CT 2048
#define CD 512
#define CH 8
#define CHS 64
#define CM (CB*CT)                 // 8192 rows

static __device__ __forceinline__ v8f wmma_f16(const FragH& a, const FragH& b, v8f c) {
    return __builtin_amdgcn_wmma_f32_16x16x32_f16(
        /*neg_a=*/false, a.h, /*neg_b=*/false, b.h,
        /*c_mod=*/(short)0, c, /*reuse_a=*/false, /*reuse_b=*/false);
}

// ---------------------------------------------------------------------------
// f32 -> f16 conversion (weights)
// ---------------------------------------------------------------------------
__global__ void __launch_bounds__(256) cvt_f32_to_f16(const float* __restrict__ src,
                                                      _Float16* __restrict__ dst, int n) {
    int i = blockIdx.x * 256 + threadIdx.x;
    if (i < n) dst[i] = (_Float16)src[i];
}

// ---------------------------------------------------------------------------
// LayerNorm over rows of length 512, f32 in, f16 out. One block per row.
// ---------------------------------------------------------------------------
__global__ void __launch_bounds__(256) ln_kernel(const float* __restrict__ x,
                                                 const float* __restrict__ g,
                                                 const float* __restrict__ be,
                                                 _Float16* __restrict__ out) {
    __shared__ float red[256];
    const int row = blockIdx.x;
    const int tid = threadIdx.x;
    const float* xr = x + (size_t)row * CD;
    float v0 = xr[tid], v1 = xr[tid + 256];
    red[tid] = v0 + v1;
    __syncthreads();
    for (int s = 128; s > 0; s >>= 1) {
        if (tid < s) red[tid] += red[tid + s];
        __syncthreads();
    }
    float mu = red[0] * (1.0f / CD);
    __syncthreads();
    float d0 = v0 - mu, d1 = v1 - mu;
    red[tid] = d0 * d0 + d1 * d1;
    __syncthreads();
    for (int s = 128; s > 0; s >>= 1) {
        if (tid < s) red[tid] += red[tid + s];
        __syncthreads();
    }
    float rstd = rsqrtf(red[0] * (1.0f / CD) + 1e-5f);
    _Float16* orow = out + (size_t)row * CD;
    orow[tid]       = (_Float16)(d0 * rstd * g[tid] + be[tid]);
    orow[tid + 256] = (_Float16)(d1 * rstd * g[tid + 256] + be[tid + 256]);
}

// ---------------------------------------------------------------------------
// QKV projection. grid = (M/256, H, 3[q,k,v]); block = 256 (8 waves).
// Each wave: 32 rows x 64 cols (= full head width), K=512.
// Q scaled by HS^-0.5 at store; K stored TRANSPOSED as [B,H,HS,T] so that the
// score GEMM's B fragments are contiguous loads (and the transposed store is
// itself a contiguous 16B packed store per lane, thanks to the WMMA D layout).
// ---------------------------------------------------------------------------
__global__ void __launch_bounds__(256) qkv_kernel(const _Float16* __restrict__ hn,
                                                  const _Float16* __restrict__ Wq16,
                                                  const _Float16* __restrict__ Wk16,
                                                  const _Float16* __restrict__ Wv16,
                                                  _Float16* __restrict__ q16,
                                                  _Float16* __restrict__ kT16,
                                                  _Float16* __restrict__ v16p) {
    const int lane = threadIdx.x & 31;
    const int wid  = threadIdx.x >> 5;
    const int r15  = lane & 15;
    const int hi   = lane >> 4;
    const int h    = blockIdx.y;
    const int z    = blockIdx.z;
    const _Float16* Bw = (z == 0 ? Wq16 : (z == 1 ? Wk16 : Wv16)) + (size_t)h * CD * CHS;
    const int rowbase = blockIdx.x * 256 + wid * 32;

    v8f acc[2][4];
#pragma unroll
    for (int s = 0; s < 2; ++s)
#pragma unroll
        for (int j = 0; j < 4; ++j) acc[s][j] = (v8f)0.0f;

    for (int kb = 0; kb < CD; kb += 32) {
        FragH a[2];
#pragma unroll
        for (int s = 0; s < 2; ++s) {
            const _Float16* ap = hn + (size_t)(rowbase + s * 16 + r15) * CD + kb + 8 * hi;
            a[s].q[0] = *reinterpret_cast<const uint4*>(ap);
            a[s].q[1] = *reinterpret_cast<const uint4*>(ap + 16);
        }
        FragH bf[4];
#pragma unroll
        for (int j = 0; j < 4; ++j) {
            const _Float16* bp = Bw + (size_t)kb * CHS + j * 16 + r15;
#pragma unroll
            for (int e = 0; e < 16; ++e)
                bf[j].e[e] = bp[(size_t)((e & 7) + 8 * hi + 16 * (e >> 3)) * CHS];
        }
#pragma unroll
        for (int s = 0; s < 2; ++s)
#pragma unroll
            for (int j = 0; j < 4; ++j) acc[s][j] = wmma_f16(a[s], bf[j], acc[s][j]);
    }

    if (z == 1) {
        // K transposed store: kT[b, h, n, t], t contiguous per lane.
#pragma unroll
        for (int s = 0; s < 2; ++s) {
#pragma unroll
            for (int j = 0; j < 4; ++j) {
                const int n  = j * 16 + r15;
                const int tb = rowbase + s * 16 + 8 * hi;
                const int b  = tb >> 11;
                const int t  = tb & (CT - 1);
                union { _Float16 e[8]; uint4 q; } pk;
#pragma unroll
                for (int r = 0; r < 8; ++r) pk.e[r] = (_Float16)acc[s][j][r];
                *reinterpret_cast<uint4*>(kT16 + ((size_t)(b * CH + h) * CHS + n) * CT + t) = pk.q;
            }
        }
    } else {
        _Float16* dst = (z == 0) ? q16 : v16p;
        const float scale = (z == 0) ? 0.125f : 1.0f;  // HS^-0.5 = 1/8
#pragma unroll
        for (int s = 0; s < 2; ++s)
#pragma unroll
            for (int j = 0; j < 4; ++j) {
                const int n = j * 16 + r15;
#pragma unroll
                for (int r = 0; r < 8; ++r) {
                    const int m = rowbase + s * 16 + 8 * hi + r;
                    const int b = m >> 11;
                    const int t = m & (CT - 1);
                    dst[((size_t)(b * CH + h) * CT + t) * CHS + n] =
                        (_Float16)(acc[s][j][r] * scale);
                }
            }
    }
}

// ---------------------------------------------------------------------------
// Fused attention: scores(WMMA)+ALiBi -> LDS(f32) -> softmax -> NT-store probs
// -> P@V (WMMA, probs re-read from LDS as f16) -> attn output (f16, head-concat)
// grid = (T/16, H, B); block = 256 (8 waves). Dynamic LDS = 135168 bytes.
// ---------------------------------------------------------------------------
__global__ void __launch_bounds__(256) attn_kernel(const _Float16* __restrict__ q16,
                                                   const _Float16* __restrict__ kT16,
                                                   const _Float16* __restrict__ v16p,
                                                   float* __restrict__ gprobs,
                                                   _Float16* __restrict__ attn16) {
    extern __shared__ float smem[];
    float* sc = smem;                 // [16][2048] scores -> probs
    float* pv = smem + 16 * CT;       // [16][64]  P@V accumulator

    const int tid  = threadIdx.x;
    const int lane = tid & 31;
    const int wid  = tid >> 5;
    const int r15  = lane & 15;
    const int hi   = lane >> 4;
    const int t0   = blockIdx.x * 16;
    const int h    = blockIdx.y;
    const int b    = blockIdx.z;
    const int bh   = b * CH + h;

    for (int i = tid; i < 16 * CHS; i += 256) pv[i] = 0.0f;

    // ---- Phase 1: S = Q * K^T + alibi  (per wave: 16 rows x 256 cols) ----
    const _Float16* qbase = q16 + ((size_t)bh * CT + t0 + r15) * CHS;
    FragH a0, a1;
    a0.q[0] = *reinterpret_cast<const uint4*>(qbase + 8 * hi);
    a0.q[1] = *reinterpret_cast<const uint4*>(qbase + 16 + 8 * hi);
    a1.q[0] = *reinterpret_cast<const uint4*>(qbase + 32 + 8 * hi);
    a1.q[1] = *reinterpret_cast<const uint4*>(qbase + 48 + 8 * hi);
    const float slope = 0.5f * (float)(h + 1);   // 2^(-8/H) * (h+1), H=8
    const _Float16* kbase = kT16 + (size_t)bh * CHS * CT;

    for (int j = 0; j < 16; ++j) {
        const int n0 = wid * 256 + j * 16;
        FragH b0, b1;
        const _Float16* bp0 = kbase + n0 + r15;
        const _Float16* bp1 = kbase + (size_t)32 * CT + n0 + r15;
#pragma unroll
        for (int e = 0; e < 16; ++e) {
            const size_t ko = (size_t)((e & 7) + 8 * hi + 16 * (e >> 3)) * CT;
            b0.e[e] = bp0[ko];
            b1.e[e] = bp1[ko];
        }
        v8f s = (v8f)0.0f;
        s = wmma_f16(a0, b0, s);
        s = wmma_f16(a1, b1, s);
        const int scol = n0 + r15;
        const float ab = -slope * (float)scol;
#pragma unroll
        for (int r = 0; r < 8; ++r) sc[(size_t)(r + 8 * hi) * CT + scol] = s[r] + ab;
    }
    __syncthreads();

    // ---- Phase 2: per-row softmax (wave handles 2 full rows) + NT store ----
    for (int rr = 0; rr < 2; ++rr) {
        const int m = wid * 2 + rr;
        const int t = t0 + m;
        float* row = sc + (size_t)m * CT;
        float lmax = -1e30f;
        for (int i = lane; i < CT; i += 32) lmax = fmaxf(lmax, row[i]);
#pragma unroll
        for (int off = 16; off >= 1; off >>= 1) lmax = fmaxf(lmax, __shfl_xor(lmax, off, 32));
        float lsum = 0.0f;
        for (int i = lane; i < CT; i += 32) {
            float e = __expf(row[i] - lmax);
            row[i] = e;
            lsum += e;
        }
#pragma unroll
        for (int off = 16; off >= 1; off >>= 1) lsum += __shfl_xor(lsum, off, 32);
        const float rinv = 1.0f / lsum;
        float* gp = gprobs + ((size_t)(h * CB + b) * CT + t) * CT;   // [H,B,T,T]
        for (int i = lane; i < CT; i += 32) {
            float p = row[i] * rinv;
            row[i] = p;
            __builtin_nontemporal_store(p, gp + i);                  // TH=NT stream-out
        }
    }
    __syncthreads();

    // ---- Phase 3: O += P * V  (wave reduces its 256-wide K slice) ----
    const _Float16* vbase = v16p + (size_t)bh * CT * CHS;
    v8f accv[4];
#pragma unroll
    for (int j = 0; j < 4; ++j) accv[j] = (v8f)0.0f;
    for (int step = 0; step < 8; ++step) {
        const int kb2 = wid * 256 + step * 32;
        FragH pa;
        const float* sr = sc + (size_t)r15 * CT + kb2 + 8 * hi;
#pragma unroll
        for (int jj = 0; jj < 8; ++jj) pa.e[jj] = (_Float16)sr[jj];
#pragma unroll
        for (int jj = 0; jj < 8; ++jj) pa.e[8 + jj] = (_Float16)sr[16 + jj];
#pragma unroll
        for (int j = 0; j < 4; ++j) {
            FragH vb;
            const _Float16* vp = vbase + (size_t)kb2 * CHS + j * 16 + r15;
#pragma unroll
            for (int e = 0; e < 16; ++e)
                vb.e[e] = vp[(size_t)((e & 7) + 8 * hi + 16 * (e >> 3)) * CHS];
            accv[j] = wmma_f16(pa, vb, accv[j]);
        }
    }
#pragma unroll
    for (int j = 0; j < 4; ++j) {
        const int n = j * 16 + r15;
#pragma unroll
        for (int r = 0; r < 8; ++r)
            atomicAdd(&pv[(size_t)(r + 8 * hi) * CHS + n], accv[j][r]);  // ds_add_f32
    }
    __syncthreads();

    // ---- Phase 4: write head-concatenated output [B,T,D] f16 ----
    for (int i = tid; i < 16 * CHS; i += 256) {
        const int m = i >> 6;
        const int n = i & 63;
        attn16[((size_t)(b * CT + t0 + m)) * CD + h * CHS + n] = (_Float16)pv[i];
    }
}

// ---------------------------------------------------------------------------
// Generic f16 WMMA GEMM: C[MxN] = A[MxK]*Bw[KxN] + bias (+res) (ReLU) ->f16/f32
// block = 256 (8 waves, 2x4); block tile 64x256; wave tile 32x64.
// grid = (M/64, N/256).
// ---------------------------------------------------------------------------
template <bool RELU, bool OUT_F16, bool HAS_RES>
__global__ void __launch_bounds__(256) gemm_kernel(const _Float16* __restrict__ A, int lda,
                                                   const _Float16* __restrict__ Bw, int ldb,
                                                   const float* __restrict__ bias,
                                                   const float* __restrict__ res,
                                                   void* __restrict__ Cout,
                                                   int M, int N, int K) {
    const int lane = threadIdx.x & 31;
    const int wid  = threadIdx.x >> 5;
    const int wm   = wid >> 2;
    const int wn   = wid & 3;
    const int r15  = lane & 15;
    const int hi   = lane >> 4;
    const int rowbase = blockIdx.x * 64 + wm * 32;
    const int colbase = blockIdx.y * 256 + wn * 64;

    v8f acc[2][4];
#pragma unroll
    for (int s = 0; s < 2; ++s)
#pragma unroll
        for (int j = 0; j < 4; ++j) acc[s][j] = (v8f)0.0f;

    for (int kb = 0; kb < K; kb += 32) {
        if (kb + 32 < K)   // hint next B rows into cache: global_prefetch_b8
            __builtin_prefetch(Bw + (size_t)(kb + 32 + lane) * ldb + colbase, 0, 1);
        FragH a[2];
#pragma unroll
        for (int s = 0; s < 2; ++s) {
            const _Float16* ap = A + (size_t)(rowbase + s * 16 + r15) * lda + kb + 8 * hi;
            a[s].q[0] = *reinterpret_cast<const uint4*>(ap);
            a[s].q[1] = *reinterpret_cast<const uint4*>(ap + 16);
        }
        FragH bf[4];
#pragma unroll
        for (int j = 0; j < 4; ++j) {
            const _Float16* bp = Bw + (size_t)kb * ldb + colbase + j * 16 + r15;
#pragma unroll
            for (int e = 0; e < 16; ++e)
                bf[j].e[e] = bp[(size_t)((e & 7) + 8 * hi + 16 * (e >> 3)) * ldb];
        }
#pragma unroll
        for (int s = 0; s < 2; ++s)
#pragma unroll
            for (int j = 0; j < 4; ++j) acc[s][j] = wmma_f16(a[s], bf[j], acc[s][j]);
    }

#pragma unroll
    for (int j = 0; j < 4; ++j) {
        const int n = colbase + j * 16 + r15;
        const float bv = bias[n];
#pragma unroll
        for (int s = 0; s < 2; ++s)
#pragma unroll
            for (int r = 0; r < 8; ++r) {
                const int m = rowbase + s * 16 + 8 * hi + r;
                float v = acc[s][j][r] + bv;
                if (HAS_RES) v += res[(size_t)m * N + n];
                if (RELU) v = fmaxf(v, 0.0f);
                if (OUT_F16)
                    reinterpret_cast<_Float16*>(Cout)[(size_t)m * N + n] = (_Float16)v;
                else
                    reinterpret_cast<float*>(Cout)[(size_t)m * N + n] = v;
            }
    }
}

// ---------------------------------------------------------------------------
// Host launcher
// ---------------------------------------------------------------------------
extern "C" void kernel_launch(void* const* d_in, const int* in_sizes, int n_in,
                              void* d_out, int out_size, void* d_ws, size_t ws_size,
                              hipStream_t stream) {
    (void)in_sizes; (void)n_in; (void)out_size; (void)ws_size;
    const float* x   = (const float*)d_in[0];
    const float* Wq  = (const float*)d_in[1];
    const float* Wk  = (const float*)d_in[2];
    const float* Wv  = (const float*)d_in[3];
    const float* Wp  = (const float*)d_in[4];
    const float* bp  = (const float*)d_in[5];
    const float* W1  = (const float*)d_in[6];
    const float* b1  = (const float*)d_in[7];
    const float* W2  = (const float*)d_in[8];
    const float* b2  = (const float*)d_in[9];
    const float* g1  = (const float*)d_in[10];
    const float* be1 = (const float*)d_in[11];
    const float* g2  = (const float*)d_in[12];
    const float* be2 = (const float*)d_in[13];

    char* ws = (char*)d_ws;
    size_t off = 0;
    auto carve = [&](size_t bytes) { char* p = ws + off; off += (bytes + 255) & ~size_t(255); return p; };
    _Float16* hn16   = (_Float16*)carve((size_t)CM * CD * 2);
    _Float16* q16    = (_Float16*)carve((size_t)CM * CD * 2);
    _Float16* kT16   = (_Float16*)carve((size_t)CM * CD * 2);
    _Float16* v16p   = (_Float16*)carve((size_t)CM * CD * 2);
    _Float16* attn16 = (_Float16*)carve((size_t)CM * CD * 2);
    _Float16* h2_16  = (_Float16*)carve((size_t)CM * CD * 2);
    float*    x2f    = (float*)   carve((size_t)CM * CD * 4);
    _Float16* ff16   = (_Float16*)carve((size_t)CM * 4 * CD * 2);
    _Float16* Wq16   = (_Float16*)carve((size_t)CH * CD * CHS * 2);
    _Float16* Wk16   = (_Float16*)carve((size_t)CH * CD * CHS * 2);
    _Float16* Wv16   = (_Float16*)carve((size_t)CH * CD * CHS * 2);
    _Float16* Wp16   = (_Float16*)carve((size_t)CD * CD * 2);
    _Float16* W116   = (_Float16*)carve((size_t)CD * 4 * CD * 2);
    _Float16* W216   = (_Float16*)carve((size_t)4 * CD * CD * 2);

    float* out_x     = (float*)d_out;                       // [B,T,D]
    float* out_probs = (float*)d_out + (size_t)CM * CD;     // [H,B,T,T]

    // 1) weight conversion
    const int nWh = CH * CD * CHS;      // 262144
    const int nWp = CD * CD;            // 262144
    const int nW1 = CD * 4 * CD;        // 1048576
    cvt_f32_to_f16<<<(nWh + 255) / 256, 256, 0, stream>>>(Wq, Wq16, nWh);
    cvt_f32_to_f16<<<(nWh + 255) / 256, 256, 0, stream>>>(Wk, Wk16, nWh);
    cvt_f32_to_f16<<<(nWh + 255) / 256, 256, 0, stream>>>(Wv, Wv16, nWh);
    cvt_f32_to_f16<<<(nWp + 255) / 256, 256, 0, stream>>>(Wp, Wp16, nWp);
    cvt_f32_to_f16<<<(nW1 + 255) / 256, 256, 0, stream>>>(W1, W116, nW1);
    cvt_f32_to_f16<<<(nW1 + 255) / 256, 256, 0, stream>>>(W2, W216, nW1);

    // 2) LN1
    ln_kernel<<<CM, 256, 0, stream>>>(x, g1, be1, hn16);

    // 3) QKV projections (Q scaled, K transposed)
    qkv_kernel<<<dim3(CM / 256, CH, 3), 256, 0, stream>>>(hn16, Wq16, Wk16, Wv16,
                                                          q16, kT16, v16p);

    // 4) fused attention
    const int SMEM = (16 * CT + 16 * CHS) * sizeof(float);  // 135168 B
    hipFuncSetAttribute((const void*)attn_kernel,
                        hipFuncAttributeMaxDynamicSharedMemorySize, SMEM);
    attn_kernel<<<dim3(CT / 16, CH, CB), 256, SMEM, stream>>>(q16, kT16, v16p,
                                                              out_probs, attn16);

    // 5) output projection + residual -> x2 (f32)
    gemm_kernel<false, false, true><<<dim3(CM / 64, CD / 256), 256, 0, stream>>>(
        attn16, CD, Wp16, CD, bp, x, x2f, CM, CD, CD);

    // 6) LN2
    ln_kernel<<<CM, 256, 0, stream>>>(x2f, g2, be2, h2_16);

    // 7) FFN up + ReLU (f16)
    gemm_kernel<true, true, false><<<dim3(CM / 64, (4 * CD) / 256), 256, 0, stream>>>(
        h2_16, CD, W116, 4 * CD, b1, nullptr, ff16, CM, 4 * CD, CD);

    // 8) FFN down + residual -> final x (f32, d_out)
    gemm_kernel<false, false, true><<<dim3(CM / 64, CD / 256), 256, 0, stream>>>(
        ff16, 4 * CD, W216, CD, b2, x2f, out_x, CM, CD, 4 * CD);
}
